// Head_21809843929106
// MI455X (gfx1250) — compile-verified, matrix operand order checked by reading, not run on hardware
//
#include <hip/hip_runtime.h>
#include <hip/hip_bf16.h>
#include <stdint.h>

typedef __attribute__((ext_vector_type(16))) __bf16 v16bf;
typedef __attribute__((ext_vector_type(8)))  float  v8f;
typedef __attribute__((ext_vector_type(4)))  unsigned v4u;
typedef __attribute__((ext_vector_type(8)))  int    v8i;
typedef __attribute__((ext_vector_type(4)))  int    v4i;

#define EMBED 1024
#define NHCOL 320     // Q1|K1|Q2|K2|V column blocks of 64
#define HEADD 64
#define TSEQ  2048
#define NROWS 8192    // B*T

#if __has_builtin(__builtin_amdgcn_tensor_load_to_lds) && __has_builtin(__builtin_amdgcn_s_wait_tensorcnt)
#define HAVE_TDM 1
#endif

// ---------- helpers ----------

__device__ __forceinline__ unsigned short f2bf(float f) {
  union { float f; unsigned u; } v; v.f = f;
  unsigned r = v.u + 0x7FFFu + ((v.u >> 16) & 1u);  // round-to-nearest-even
  return (unsigned short)(r >> 16);
}

// Load a 16x32 bf16 fragment (A layout; also B layout when src is stored
// N-major with K contiguous). ISA 7.12.2: lane l<16 -> row l, K {0..7,16..23};
// lane l+16 -> row l, K {8..15,24..31}.  Two 16B loads per lane.
__device__ __forceinline__ v16bf ldfrag(const unsigned short* src, int stride) {
  int l  = threadIdx.x & 31;
  int r  = l & 15;
  int hf = l >> 4;
  union { v16bf v; uint4 q[2]; } u;
  const unsigned short* p = src + r * stride + hf * 8;
  u.q[0] = *(const uint4*)(p);
  u.q[1] = *(const uint4*)(p + 16);
  return u.v;
}

__device__ __forceinline__ v8f wmma_bf16(v16bf a, v16bf b, v8f c) {
  return __builtin_amdgcn_wmma_f32_16x16x32_bf16(false, a, false, b,
                                                 (short)0, c, false, false);
}

#ifdef HAVE_TDM
// Issue one TDM 2-D tile load: 32 rows x 64 bf16 cols from proj (row stride
// NHCOL elements) into LDS at ldsOff, row-major.  D# layout per ISA ch.8.
__device__ __forceinline__ void tdm_load_tile(const unsigned short* gsrc, unsigned ldsOff) {
  unsigned long long ga = (unsigned long long)(uintptr_t)gsrc;
  v4u g0;
  g0[0] = 1u;                                   // count=1, user mode, no gather
  g0[1] = ldsOff;                               // lds_addr (bytes)
  g0[2] = (unsigned)ga;                         // global_addr[31:0]
  g0[3] = (unsigned)(ga >> 32) | 0x80000000u;   // global_addr[56:32] | type=2
  v8i g1;
  g1[0] = (int)(1u << 16);                      // data_size = 2 bytes
  g1[1] = (int)((unsigned)NHCOL << 16);         // tensor_dim0[15:0] = 320
  g1[2] = (int)((unsigned)NROWS << 16);         // tensor_dim1[15:0] = 8192
  g1[3] = (int)(64u << 16);                     // tile_dim0 = 64
  g1[4] = 32;                                   // tile_dim1 = 32
  g1[5] = NHCOL;                                // tensor_dim0_stride = 320
  g1[6] = 0;
  g1[7] = 0;
  v4i z4 = {};
  v8i z8 = {};
  __builtin_amdgcn_tensor_load_to_lds(g0, g1, z4, z4, z8, 0);
}
#endif

// ---------- kernel 1: x f32 -> bf16 ----------

__global__ void k_convx(const float* __restrict__ x,
                        unsigned short* __restrict__ xbf, int n4) {
  int i = blockIdx.x * blockDim.x + threadIdx.x;
  if (i >= n4) return;
  float4 f = ((const float4*)x)[i];
  union { unsigned short s[4]; uint2 u; } o;
  o.s[0] = f2bf(f.x); o.s[1] = f2bf(f.y); o.s[2] = f2bf(f.z); o.s[3] = f2bf(f.w);
  ((uint2*)xbf)[i] = o.u;
}

// ---------- kernel 2: build transposed packed weights Wt[320][1024] bf16 ----------
// col blocks: 0-63 Q1*0.125 | 64-127 K1 | 128-191 Q2*0.125 | 192-255 K2 | 256-319 V

__global__ void k_convw(const float* __restrict__ Wk1, const float* __restrict__ Wq1,
                        const float* __restrict__ Wk2, const float* __restrict__ Wq2,
                        const float* __restrict__ Wv,  unsigned short* __restrict__ Wt) {
  int idx = blockIdx.x * blockDim.x + threadIdx.x;
  if (idx >= NHCOL * EMBED) return;
  int col = idx >> 10;
  int k   = idx & (EMBED - 1);
  int blk = col >> 6;
  int h   = col & 63;
  const float* W; float s = 1.0f;
  if (blk == 0)      { W = Wq1; s = 0.125f; }
  else if (blk == 1) { W = Wk1; }
  else if (blk == 2) { W = Wq2; s = 0.125f; }
  else if (blk == 3) { W = Wk2; }
  else               { W = Wv;  }
  Wt[idx] = f2bf(W[(size_t)k * HEADD + h] * s);
}

// ---------- kernel 3: projection GEMM (8192x1024)x(1024x320) -> proj bf16 ----------

__global__ __launch_bounds__(32) void k_proj(const unsigned short* __restrict__ xbf,
                                             const unsigned short* __restrict__ Wt,
                                             unsigned short* __restrict__ proj) {
  int m0 = blockIdx.x * 16;
  int n0 = blockIdx.y * 16;
  const unsigned short* a0 = xbf + (size_t)m0 * EMBED;
  const unsigned short* b0 = Wt  + (size_t)n0 * EMBED;
  v8f c = {};
#pragma unroll 4
  for (int k = 0; k < EMBED; k += 32)
    c = wmma_bf16(ldfrag(a0 + k, EMBED), ldfrag(b0 + k, EMBED), c);

  int l    = threadIdx.x & 31;
  int col  = n0 + (l & 15);
  int row0 = m0 + ((l >> 4) << 3);
#pragma unroll
  for (int r = 0; r < 8; ++r)
    proj[(size_t)(row0 + r) * NHCOL + col] = f2bf(c[r]);
}

// ---------- kernel 4: fused differential flash attention ----------
// Block = 128 q rows (8 waves x 16).  Streams 32-key chunks through LDS,
// staged by the Tensor Data Mover when available.

__global__ __launch_bounds__(256) void k_attn(const unsigned short* __restrict__ proj,
                                              const float* __restrict__ lambp,
                                              float* __restrict__ out) {
  __shared__ __align__(16) unsigned short K1s[32 * 64];   // [key][h]
  __shared__ __align__(16) unsigned short K2s[32 * 64];   // [key][h]
  __shared__ __align__(16) unsigned short Vts[64 * 32];   // [h][key] (transposed)
  __shared__ __align__(16) unsigned short P1s[8][16 * 32];
  __shared__ __align__(16) unsigned short P2s[8][16 * 32];
#ifdef HAVE_TDM
  __shared__ __align__(16) unsigned short Vss[32 * 64];   // TDM staging [key][h]
#endif

  const int b    = blockIdx.x >> 4;
  const int qb   = blockIdx.x & 15;
  const int wave = threadIdx.x >> 5;
  const int lane = threadIdx.x & 31;
  const int lr   = lane & 15;
  const int half = lane >> 4;

  const float lamb  = lambp[0];
  const int   qrow0 = b * TSEQ + qb * 128 + wave * 16;
  const unsigned short* qbase = proj + (size_t)qrow0 * NHCOL;

  v16bf q1lo = ldfrag(qbase +   0, NHCOL);
  v16bf q1hi = ldfrag(qbase +  32, NHCOL);
  v16bf q2lo = ldfrag(qbase + 128, NHCOL);
  v16bf q2hi = ldfrag(qbase + 160, NHCOL);

  v8f O1[4] = {}, O2[4] = {};
  float m1[8], l1[8], m2[8], l2[8];
#pragma unroll
  for (int r = 0; r < 8; ++r) { m1[r] = -1e30f; m2[r] = -1e30f; l1[r] = 0.f; l2[r] = 0.f; }

  const int tt  = threadIdx.x;
  const int key = tt >> 3;   // 0..31
  const int seg = tt & 7;    // 16B segment along h

  unsigned short* P1w = P1s[wave];
  unsigned short* P2w = P2s[wave];

  for (int kc = 0; kc < TSEQ / 32; ++kc) {
    const int krow0 = b * TSEQ + kc * 32;

#ifdef HAVE_TDM
    if (wave == 0) {
      const unsigned short* rowp = proj + (size_t)krow0 * NHCOL;
      tdm_load_tile(rowp +  64, (unsigned)(uintptr_t)(void*)K1s);
      tdm_load_tile(rowp + 192, (unsigned)(uintptr_t)(void*)K2s);
      tdm_load_tile(rowp + 256, (unsigned)(uintptr_t)(void*)Vss);
      __builtin_amdgcn_s_wait_tensorcnt(0);
    }
    __syncthreads();                 // tiles in LDS, visible to all waves
    {                                // LDS->LDS transpose of V
      union { uint4 q; unsigned short s[8]; } vv;
      vv.q = ((const uint4*)Vss)[key * 8 + seg];
#pragma unroll
      for (int i = 0; i < 8; ++i)
        Vts[(seg * 8 + i) * 32 + key] = vv.s[i];
    }
    if (kc + 1 < TSEQ / 32)
      __builtin_prefetch(proj + (size_t)(krow0 + 32 + key) * NHCOL + 64, 0, 0);
    __syncthreads();                 // Vts ready
#else
    const unsigned short* kr = proj + (size_t)(krow0 + key) * NHCOL;
    ((uint4*)K1s)[key * 8 + seg] = *(const uint4*)(kr +  64 + seg * 8);
    ((uint4*)K2s)[key * 8 + seg] = *(const uint4*)(kr + 192 + seg * 8);
    union { uint4 q; unsigned short s[8]; } vv;
    vv.q = *(const uint4*)(kr + 256 + seg * 8);
#pragma unroll
    for (int i = 0; i < 8; ++i)
      Vts[(seg * 8 + i) * 32 + key] = vv.s[i];
    if (kc + 1 < TSEQ / 32)
      __builtin_prefetch(proj + (size_t)(krow0 + 32 + key) * NHCOL + 64, 0, 0);
    __syncthreads();
#endif

    // S tiles: 16q x 32keys per stream
    v8f s1a = {}, s1b = {}, s2a = {}, s2b = {};
    s1a = wmma_bf16(q1lo, ldfrag(K1s,                64), s1a);
    s1a = wmma_bf16(q1hi, ldfrag(K1s + 32,           64), s1a);
    s1b = wmma_bf16(q1lo, ldfrag(K1s + 16 * 64,      64), s1b);
    s1b = wmma_bf16(q1hi, ldfrag(K1s + 16 * 64 + 32, 64), s1b);
    s2a = wmma_bf16(q2lo, ldfrag(K2s,                64), s2a);
    s2a = wmma_bf16(q2hi, ldfrag(K2s + 32,           64), s2a);
    s2b = wmma_bf16(q2lo, ldfrag(K2s + 16 * 64,      64), s2b);
    s2b = wmma_bf16(q2hi, ldfrag(K2s + 16 * 64 + 32, 64), s2b);

    // online softmax, both streams (row m lives across 16-lane halves)
    float c1[8], c2[8];
#pragma unroll
    for (int r = 0; r < 8; ++r) {
      int row = r + half * 8;
      // stream 1
      float t = fmaxf(s1a[r], s1b[r]);
      t = fmaxf(t, __shfl_xor(t, 8, 16));
      t = fmaxf(t, __shfl_xor(t, 4, 16));
      t = fmaxf(t, __shfl_xor(t, 2, 16));
      t = fmaxf(t, __shfl_xor(t, 1, 16));
      float mn = fmaxf(m1[r], t);
      float cr = __expf(m1[r] - mn);
      float pa = __expf(s1a[r] - mn);
      float pb = __expf(s1b[r] - mn);
      float rs = pa + pb;
      rs += __shfl_xor(rs, 8, 16);
      rs += __shfl_xor(rs, 4, 16);
      rs += __shfl_xor(rs, 2, 16);
      rs += __shfl_xor(rs, 1, 16);
      l1[r] = l1[r] * cr + rs; m1[r] = mn; c1[r] = cr;
      P1w[row * 32 + lr]      = f2bf(pa);
      P1w[row * 32 + 16 + lr] = f2bf(pb);
      // stream 2
      float u = fmaxf(s2a[r], s2b[r]);
      u = fmaxf(u, __shfl_xor(u, 8, 16));
      u = fmaxf(u, __shfl_xor(u, 4, 16));
      u = fmaxf(u, __shfl_xor(u, 2, 16));
      u = fmaxf(u, __shfl_xor(u, 1, 16));
      float mn2 = fmaxf(m2[r], u);
      float cr2 = __expf(m2[r] - mn2);
      float qa  = __expf(s2a[r] - mn2);
      float qb2 = __expf(s2b[r] - mn2);
      float rs2 = qa + qb2;
      rs2 += __shfl_xor(rs2, 8, 16);
      rs2 += __shfl_xor(rs2, 4, 16);
      rs2 += __shfl_xor(rs2, 2, 16);
      rs2 += __shfl_xor(rs2, 1, 16);
      l2[r] = l2[r] * cr2 + rs2; m2[r] = mn2; c2[r] = cr2;
      P2w[row * 32 + lr]      = f2bf(qa);
      P2w[row * 32 + 16 + lr] = f2bf(qb2);
    }
#pragma unroll
    for (int j = 0; j < 4; ++j)
#pragma unroll
      for (int r = 0; r < 8; ++r) { O1[j][r] *= c1[r]; O2[j][r] *= c2[r]; }

    // O += P * V
    v16bf p1f = ldfrag(P1w, 32);
    v16bf p2f = ldfrag(P2w, 32);
#pragma unroll
    for (int j = 0; j < 4; ++j) {
      v16bf vf = ldfrag(Vts + j * 16 * 32, 32);
      O1[j] = wmma_bf16(p1f, vf, O1[j]);
      O2[j] = wmma_bf16(p2f, vf, O2[j]);
    }
    __syncthreads();
  }

  // epilogue: out = O1/l1 - lamb*O2/l2
#pragma unroll
  for (int j = 0; j < 4; ++j)
#pragma unroll
    for (int r = 0; r < 8; ++r) {
      float o = O1[j][r] / l1[r] - lamb * (O2[j][r] / l2[r]);
      out[(size_t)(qrow0 + r + half * 8) * HEADD + j * 16 + lr] = o;
    }
}

// ---------- launch ----------

extern "C" void kernel_launch(void* const* d_in, const int* in_sizes, int n_in,
                              void* d_out, int out_size, void* d_ws, size_t ws_size,
                              hipStream_t stream) {
  const float* x    = (const float*)d_in[0];
  const float* lamb = (const float*)d_in[1];
  const float* Wk1  = (const float*)d_in[2];
  const float* Wq1  = (const float*)d_in[3];
  const float* Wk2  = (const float*)d_in[4];
  const float* Wq2  = (const float*)d_in[5];
  const float* Wv   = (const float*)d_in[6];
  float* out = (float*)d_out;

  char* ws = (char*)d_ws;
  size_t off_xbf  = 0;
  size_t off_wt   = off_xbf + (size_t)NROWS * EMBED * 2;   // 16.78 MB
  size_t off_proj = off_wt  + (size_t)NHCOL * EMBED * 2;   // +0.66 MB
  unsigned short* xbf  = (unsigned short*)(ws + off_xbf);
  unsigned short* Wt   = (unsigned short*)(ws + off_wt);
  unsigned short* proj = (unsigned short*)(ws + off_proj); // 5.24 MB

  int n4 = NROWS * EMBED / 4;
  hipLaunchKernelGGL(k_convx, dim3((n4 + 255) / 256), dim3(256), 0, stream, x, xbf, n4);
  int nw = NHCOL * EMBED;
  hipLaunchKernelGGL(k_convw, dim3((nw + 255) / 256), dim3(256), 0, stream,
                     Wk1, Wq1, Wk2, Wq2, Wv, Wt);
  hipLaunchKernelGGL(k_proj, dim3(NROWS / 16, NHCOL / 16), dim3(32), 0, stream,
                     xbf, Wt, proj);
  hipLaunchKernelGGL(k_attn, dim3(NROWS / 128), dim3(256), 0, stream,
                     proj, lamb, out);
}
